// mdlstm_38036230374106
// MI455X (gfx1250) — compile-verified, matrix operand order checked by reading, not run on hardware
//
#include <hip/hip_runtime.h>
#include <hip/hip_bf16.h>
#include <stdint.h>

// ---- types ------------------------------------------------------------
typedef __attribute__((ext_vector_type(16))) _Float16 v16h;
typedef __attribute__((ext_vector_type(8)))  float    v8f;
typedef __attribute__((ext_vector_type(4)))  float    f32x4;
typedef __attribute__((ext_vector_type(4)))  _Float16 h16x4;
typedef __attribute__((ext_vector_type(4)))  uint32_t u32x4;
typedef __attribute__((ext_vector_type(2)))  uint32_t u32x2;

#define H_IMG 32
#define W_IMG 32
#define HID   256
#define GATES 1024              // 4*HID
#define SB    ((size_t)(H_IMG*W_IMG*HID))   // per-batch stride in h/s arrays = 262144
#define A_PITCH 264             // 256+8 halfs -> 528B row stride, conflict-free b128 reads
#define B_PITCH 40              // 32+8 halfs  -> 80B row stride (20 banks), conflict-free
#define B_BUF   (256 * B_PITCH) // halfs per B double-buffer slab (20 KB)

union FragU { u32x4 u[2]; v16h h; };
union PackU { h16x4 h; u32x2 u; };

__device__ __forceinline__ float fast_sigmoid(float v) {
    return 1.0f / (1.0f + __expf(-v));
}
__device__ __forceinline__ float fast_tanh(float v) {
    float e = __expf(-2.0f * v);
    return (1.0f - e) / (1.0f + e);
}

// ---- prologue: convert w_hh (1024x256 f32, row-major [n][k]) to f16 ----
__global__ void __launch_bounds__(512)
mdlstm_cvt_w16(const float* __restrict__ whh, _Float16* __restrict__ w16) {
    int i = blockIdx.x * 512 + threadIdx.x;           // 512 blocks * 512 thr = 262144
    w16[i] = (_Float16)whh[i];
}

// ---- one anti-diagonal step -------------------------------------------
// grid = (16, n_cells): blockIdx.x -> (m tile of 32 batches) x (k tile of 64 hid)
// block = 256 threads = 8 wave32; wave tile = 16 batch x 16 k, all 4 gates.
// B tiles are DMAed per K-step by the async-to-LDS engine (ASYNCcnt),
// double-buffered, so WMMAs consume short-latency ds_load_b128 fragments.
__global__ void __launch_bounds__(256)
mdlstm_diag(const float* __restrict__ x,          // [B,1024]
            const _Float16* __restrict__ w16,     // [1024,256] f16 (= w_hh)
            const float* __restrict__ wih,        // [1024]
            const float* __restrict__ bias,       // [1024]
            float* __restrict__ s_buf,            // [B, 1024, 256]
            float* __restrict__ out,              // [B, 1024, 256]  (h)
            int d, int rmin) {
    __shared__ _Float16 Alds[32 * A_PITCH];       // f16(h_up + h_left), 32 x 256
    __shared__ _Float16 Blds[2 * B_BUF];          // 2 x (256 gate-rows x 32 K)

    const int r = rmin + blockIdx.y;
    const int c = d - r;
    const int pix  = r * W_IMG + c;
    const int pixU = pix - W_IMG;
    const int pixL = pix - 1;
    const bool hasUp   = (r > 0);
    const bool hasLeft = (c > 0);

    const int mb0 = (blockIdx.x & 3) * 32;     // batch block offset
    const int kb0 = (blockIdx.x >> 2) * 64;    // hidden-index block offset

    // ---- async B staging setup: thread t owns gate-row t of the tile ----
    // local row t = g*64 + j  ->  w16 row N = g*256 + (kb0 + j); 64B per step
    const int t = threadIdx.x;
    const _Float16* gsrc = w16 + (size_t)((t >> 6) * HID + kb0 + (t & 63)) * HID;
    const uint32_t ldsB0 = (uint32_t)(uintptr_t)&Blds[t * B_PITCH];
    const uint32_t ldsB1 = (uint32_t)(uintptr_t)&Blds[B_BUF + t * B_PITCH];

#define ISSUE_B(ldsDst, kk)                                                     \
    do { const _Float16* _g = gsrc + (kk);                                      \
         asm volatile("global_load_async_to_lds_b128 %0, %1, off"               \
                      :: "v"(ldsDst), "v"(_g) : "memory");                      \
         asm volatile("global_load_async_to_lds_b128 %0, %1, off offset:16"     \
                      :: "v"(ldsDst), "v"(_g) : "memory");                      \
         asm volatile("global_load_async_to_lds_b128 %0, %1, off offset:32"     \
                      :: "v"(ldsDst), "v"(_g) : "memory");                      \
         asm volatile("global_load_async_to_lds_b128 %0, %1, off offset:48"     \
                      :: "v"(ldsDst), "v"(_g) : "memory"); } while (0)
#define WAIT_ASYNC() asm volatile("s_wait_asynccnt 0x0" ::: "memory")

    // kick off K-step 0 into buffer 0 before doing the A staging ALU work
    ISSUE_B(ldsB0, 0);

    // ---- stage A = f16(h_up + h_left), rows mb0..mb0+31, all 256 k ----
    {
        const int row  = t >> 3;                     // 0..31
        const int cseg = (t & 7) * 32;               // 0..224
        const int b    = mb0 + row;
        const float* pu = out + (size_t)b * SB + (size_t)pixU * HID + cseg;
        const float* pl = out + (size_t)b * SB + (size_t)pixL * HID + cseg;
        const f32x4 z = {0.f, 0.f, 0.f, 0.f};
        #pragma unroll
        for (int j = 0; j < 32; j += 4) {
            f32x4 a  = hasUp   ? *(const f32x4*)(pu + j) : z;
            f32x4 bl = hasLeft ? *(const f32x4*)(pl + j) : z;
            f32x4 s  = a + bl;
            PackU p;
            p.h = (h16x4){(_Float16)s.x, (_Float16)s.y, (_Float16)s.z, (_Float16)s.w};
            *(u32x2*)&Alds[row * A_PITCH + cseg + j] = p.u;
        }
    }
    WAIT_ASYNC();          // our async writes for step 0 have landed
    __syncthreads();       // ... and everyone else's, plus the A tile

    // ---- WMMA K-loop: D[16x16] per gate, K = 256 in steps of 32 ----
    const int lane  = t & 31;
    const int wave  = t >> 5;             // 0..7
    const int mw    = wave & 1;           // M subtile (0/16)
    const int kw    = wave >> 1;          // k subtile (0..3)
    const int lid   = lane & 15;
    const int ohalf = (lane >> 4) * 8;    // half-wave K offset per 16-bit layout

    const int kcol = kb0 + kw * 16 + lid;             // hidden index 0..255
    const int arow = (mw * 16 + lid) * A_PITCH;
    const int brow = kw * 16 + lid;                   // j within tile; gate g at +g*64

    v8f accI = {}, accF = {}, accG = {}, accO = {};

#define LOAD_A(f, kk)                                           \
    do { int _i = arow + (kk) + ohalf;                          \
         (f).u[0] = *(const u32x4*)&Alds[_i];                   \
         (f).u[1] = *(const u32x4*)&Alds[_i + 16]; } while (0)
#define LOAD_B(f, g, bufBase)                                   \
    do { int _i = (bufBase) + ((g) * 64 + brow) * B_PITCH + ohalf; \
         (f).u[0] = *(const u32x4*)&Blds[_i];                   \
         (f).u[1] = *(const u32x4*)&Blds[_i + 16]; } while (0)

    #pragma unroll
    for (int kk = 0; kk < HID; kk += 32) {
        const int cur  = (kk >> 5) & 1;
        const int kn   = kk + 32;
        if (kn < HID)                       // DMA next K-step into the other buffer
            ISSUE_B(cur ? ldsB0 : ldsB1, kn);

        FragU af, b0, b1, b2, b3;
        const int bufBase = cur * B_BUF;
        LOAD_A(af, kk);
        LOAD_B(b0, 0, bufBase); LOAD_B(b1, 1, bufBase);
        LOAD_B(b2, 2, bufBase); LOAD_B(b3, 3, bufBase);

        // same A operand feeds all four WMMAs: reuse-A on the first three
        accI = __builtin_amdgcn_wmma_f32_16x16x32_f16(false, af.h, false, b0.h,
                                                      (short)0, accI, true,  false);
        accF = __builtin_amdgcn_wmma_f32_16x16x32_f16(false, af.h, false, b1.h,
                                                      (short)0, accF, true,  false);
        accG = __builtin_amdgcn_wmma_f32_16x16x32_f16(false, af.h, false, b2.h,
                                                      (short)0, accG, true,  false);
        accO = __builtin_amdgcn_wmma_f32_16x16x32_f16(false, af.h, false, b3.h,
                                                      (short)0, accO, false, false);

        if (kn < HID) {    // next buffer ready: own DMA done, then block-wide
            WAIT_ASYNC();
            __syncthreads();
        }
    }
#undef LOAD_A
#undef LOAD_B
#undef ISSUE_B
#undef WAIT_ASYNC

    // ---- fused LSTM epilogue on the C layout (lane = k, vgpr = batch) ----
    const float bi = bias[kcol],         bf = bias[HID + kcol];
    const float bg = bias[2*HID + kcol], bo = bias[3*HID + kcol];
    const float wi = wih[kcol],          wf = wih[HID + kcol];
    const float wg = wih[2*HID + kcol],  wo = wih[3*HID + kcol];

    #pragma unroll
    for (int v = 0; v < 8; ++v) {
        const int b = mb0 + mw * 16 + v + ((lane >> 4) * 8);   // C layout M mapping
        const size_t bo64 = (size_t)b * SB;
        const float xp = x[(size_t)b * (H_IMG * W_IMG) + pix];

        const float su = hasUp   ? s_buf[bo64 + (size_t)pixU * HID + kcol] : 0.f;
        const float sl = hasLeft ? s_buf[bo64 + (size_t)pixL * HID + kcol] : 0.f;

        const float gi = fast_sigmoid(accI[v] + xp * wi + bi);
        const float gf = fast_sigmoid(accF[v] + xp * wf + bf);
        const float gg = fast_tanh   (accG[v] + xp * wg + bg);
        const float go = fast_sigmoid(accO[v] + xp * wo + bo);

        const float s = gf * (su + sl) + gi * gg;
        const float h = go * fast_tanh(s);

        s_buf[bo64 + (size_t)pix * HID + kcol] = s;
        out  [bo64 + (size_t)pix * HID + kcol] = h;
    }
}

// ---- host ---------------------------------------------------------------
extern "C" void kernel_launch(void* const* d_in, const int* in_sizes, int n_in,
                              void* d_out, int out_size, void* d_ws, size_t ws_size,
                              hipStream_t stream) {
    const float* x    = (const float*)d_in[0];   // [128,1024]
    const float* wih  = (const float*)d_in[1];   // [1024,1]
    const float* whh  = (const float*)d_in[2];   // [1024,256]
    const float* bias = (const float*)d_in[3];   // [1024]
    float* out = (float*)d_out;

    _Float16* w16 = (_Float16*)d_ws;                              // 512 KB
    float* s_buf  = (float*)((char*)d_ws + (size_t)GATES * HID * sizeof(_Float16));

    // 1) convert recurrent weights to f16 once per call
    mdlstm_cvt_w16<<<512, 512, 0, stream>>>(whh, w16);

    // 2) 63 wavefront steps; stream ordering provides the diagonal dependency
    for (int d = 0; d < H_IMG + W_IMG - 1; ++d) {
        int rmin = d - (W_IMG - 1); if (rmin < 0) rmin = 0;
        int rmax = d < (H_IMG - 1) ? d : (H_IMG - 1);
        int nc = rmax - rmin + 1;
        dim3 grid(16, nc);   // 16 blocks/cell: 4 batch-tiles x 4 k-tiles
        mdlstm_diag<<<grid, 256, 0, stream>>>(x, w16, wih, bias, s_buf, out, d, rmin);
    }
}
